// ScaledDotProductAttention_11639361372204
// MI455X (gfx1250) — matrix operand order repaired and checked by reading, not compiled, not run on hardware
//
#include <hip/hip_runtime.h>
#include <math.h>

// CDNA5 / gfx1250 wave32 WMMA types
typedef __attribute__((ext_vector_type(16))) __bf16 v16bf;
typedef __attribute__((ext_vector_type(8)))  float  v8f;

#define B_  4
#define H_  16
#define S_  2048
#define D_  64
#define M_  32                      // query rows per workgroup
#define NT  512                     // threads per block
#define NW  (NT / 32)               // 16 waves
#define KEYS_PER_WAVE (S_ / NW)     // 128 keys per wave

__global__ __launch_bounds__(NT)
void attn_fused_kernel(const float* __restrict__ Q,
                       const float* __restrict__ K,
                       const float* __restrict__ V,
                       const int*   __restrict__ mask,
                       float* __restrict__ out,   // [B,H,S,D]
                       float* __restrict__ attn)  // [B,H,S,S]
{
    // LDS: fp32 scores/probs for 32 query rows x all 2048 keys (+1 pad word
    // per row so strided row accesses hit distinct banks), reduction buffers,
    // and the cross-wave output accumulator.  Total ~272.6 KB (< 320 KB/WGP).
    __shared__ float P[M_][S_ + 1];      // 262,272 B
    __shared__ float red[M_][16];        // 2 KB
    __shared__ float rowstat[M_];
    __shared__ float Osh[M_][D_];        // 8 KB

    const int tid  = threadIdx.x;
    const int wave = tid >> 5;
    const int lane = tid & 31;
    const int lcol = lane & 15;          // N / M sub-index within half-wave
    const int hi   = lane >> 4;          // 0: lanes 0-15, 1: lanes 16-31

    const int bh    = blockIdx.x / (S_ / M_);
    const int mtile = blockIdx.x % (S_ / M_);
    const int m0    = mtile * M_;
    const int b     = bh / H_;

    const float* Qb = Q + (size_t)bh * S_ * D_;
    const float* Kb = K + (size_t)bh * S_ * D_;
    const float* Vb = V + (size_t)bh * S_ * D_;
    const int*   mb = mask + (size_t)b * S_;

    // ---------------- Phase 1: scores = scale * Q K^T (masked) -> LDS -------
    // A fragments for two 16-row subtiles (rows m0..m0+15 and m0+16..m0+31),
    // two k-chunks each (d 0..31, 32..63).  Reused across all key tiles.
    // 16-bit A layout: lane row = lane%16, element j -> k = 8*hi + j + (j>=8)*8
    v16bf a0lo, a1lo, a0hi, a1hi;
    {
        const float* qlo = Qb + (size_t)(m0 + lcol) * D_;
        const float* qhi = Qb + (size_t)(m0 + 16 + lcol) * D_;
        #pragma unroll
        for (int j = 0; j < 16; ++j) {
            const int k = 8 * hi + j + ((j >= 8) ? 8 : 0);
            a0lo[j] = (__bf16)qlo[k];
            a1lo[j] = (__bf16)qlo[32 + k];
            a0hi[j] = (__bf16)qhi[k];
            a1hi[j] = (__bf16)qhi[32 + k];
        }
    }

    const float scaling = 0.125f;  // rsqrt(D=64)
    #pragma unroll 1
    for (int t = 0; t < KEYS_PER_WAVE / 16; ++t) {
        const int n = wave * KEYS_PER_WAVE + t * 16 + lcol;  // this lane's key col
        const float* krow = Kb + (size_t)n * D_;
        // 16-bit B layout: col = lane%16, element j -> k = 8*hi + j + (j>=8)*8
        v16bf b0, b1;
        #pragma unroll
        for (int j = 0; j < 16; ++j) {
            const int d = 8 * hi + j + ((j >= 8) ? 8 : 0);
            b0[j] = (__bf16)krow[d];
            b1[j] = (__bf16)krow[32 + d];
        }
        v8f clo = {0.f, 0.f, 0.f, 0.f, 0.f, 0.f, 0.f, 0.f};
        v8f chi = {0.f, 0.f, 0.f, 0.f, 0.f, 0.f, 0.f, 0.f};
        clo = __builtin_amdgcn_wmma_f32_16x16x32_bf16(false, a0lo, false, b0,
                                                      (short)0, clo, false, false);
        clo = __builtin_amdgcn_wmma_f32_16x16x32_bf16(false, a1lo, false, b1,
                                                      (short)0, clo, false, false);
        chi = __builtin_amdgcn_wmma_f32_16x16x32_bf16(false, a0hi, false, b0,
                                                      (short)0, chi, false, false);
        chi = __builtin_amdgcn_wmma_f32_16x16x32_bf16(false, a1hi, false, b1,
                                                      (short)0, chi, false, false);
        const int mk = mb[n];                 // key mask (per column)
        #pragma unroll
        for (int r = 0; r < 8; ++r) {
            const int m = r + 8 * hi;         // C/D layout: row = r + 8*hi
            P[m][n]      = (mk == 1) ? -INFINITY : clo[r] * scaling;
            P[16 + m][n] = (mk == 1) ? -INFINITY : chi[r] * scaling;
        }
    }
    __syncthreads();

    // ---------------- Phase 2: row max --------------------------------------
    {
        const int r  = tid & 31;              // 32 rows
        const int ch = tid >> 5;              // 16 chunks of 128 columns
        const int c0 = ch * (S_ / 16);
        float mx = -INFINITY;
        for (int c = c0; c < c0 + S_ / 16; ++c) mx = fmaxf(mx, P[r][c]);
        red[r][ch] = mx;
    }
    __syncthreads();
    if (tid < M_) {
        float mx = -INFINITY;
        #pragma unroll
        for (int i = 0; i < 16; ++i) mx = fmaxf(mx, red[tid][i]);
        rowstat[tid] = mx;
    }
    __syncthreads();

    // ---------------- Phase 3: exp + row sum --------------------------------
    {
        const int r  = tid & 31;
        const int ch = tid >> 5;
        const int c0 = ch * (S_ / 16);
        const float rm = rowstat[r];
        float s = 0.f;
        for (int c = c0; c < c0 + S_ / 16; ++c) {
            const float p = __expf(P[r][c] - rm);   // exp(-inf)=0 for masked
            P[r][c] = p;
            s += p;
        }
        red[r][ch] = s;
    }
    __syncthreads();
    if (tid < M_) {
        float s = 0.f;
        #pragma unroll
        for (int i = 0; i < 16; ++i) s += red[tid][i];
        rowstat[tid] = 1.0f / s;
    }
    __syncthreads();

    // ------- Phase 4: normalize in LDS + stream attn_weights to HBM ---------
    {
        float* attnrow = attn + ((size_t)bh * S_ + m0) * S_;
        for (int idx = tid; idx < M_ * S_; idx += NT) {
            const int r = idx >> 11;          // idx / 2048
            const int c = idx & (S_ - 1);
            const float p = P[r][c] * rowstat[r];
            P[r][c] = p;
            attnrow[(size_t)r * S_ + c] = p;  // coalesced, written exactly once
        }
        for (int idx = tid; idx < M_ * D_; idx += NT)
            ((float*)Osh)[idx] = 0.f;
    }
    __syncthreads();

    // ---------------- Phase 5: O = P @ V (bf16 WMMA) ------------------------
    // 2 row-subtiles x 4 d-tiles of f32 accumulators (64 VGPRs per lane).
    v8f oc[2][4];
    #pragma unroll
    for (int rt = 0; rt < 2; ++rt)
        #pragma unroll
        for (int dt = 0; dt < 4; ++dt)
            oc[rt][dt] = (v8f){0.f, 0.f, 0.f, 0.f, 0.f, 0.f, 0.f, 0.f};

    const int nb = wave * KEYS_PER_WAVE;
    #pragma unroll 1
    for (int ch = 0; ch < KEYS_PER_WAVE / 32; ++ch) {   // 4 k-chunks of 32 keys
        const int nbase = nb + ch * 32;
        v16bf aplo, aphi;
        #pragma unroll
        for (int j = 0; j < 16; ++j) {
            const int k = 8 * hi + j + ((j >= 8) ? 8 : 0);
            aplo[j] = (__bf16)P[lcol][nbase + k];
            aphi[j] = (__bf16)P[16 + lcol][nbase + k];
        }
        #pragma unroll
        for (int dt = 0; dt < 4; ++dt) {
            v16bf bv;
            #pragma unroll
            for (int j = 0; j < 16; ++j) {
                const int krow = nbase + 8 * hi + j + ((j >= 8) ? 8 : 0);
                bv[j] = (__bf16)Vb[(size_t)krow * D_ + dt * 16 + lcol];
            }
            oc[0][dt] = __builtin_amdgcn_wmma_f32_16x16x32_bf16(false, aplo, false, bv,
                                                                (short)0, oc[0][dt],
                                                                false, false);
            oc[1][dt] = __builtin_amdgcn_wmma_f32_16x16x32_bf16(false, aphi, false, bv,
                                                                (short)0, oc[1][dt],
                                                                false, false);
        }
    }

    // Cross-wave reduction of partial O via LDS float atomics (ds_add_f32)
    #pragma unroll
    for (int rt = 0; rt < 2; ++rt) {
        #pragma unroll
        for (int dt = 0; dt < 4; ++dt) {
            #pragma unroll
            for (int r = 0; r < 8; ++r) {
                atomicAdd(&Osh[rt * 16 + r + 8 * hi][dt * 16 + lcol], oc[rt][dt][r]);
            }
        }
    }
    __syncthreads();

    // ---------------- Phase 6: write output ---------------------------------
    {
        float* obase = out + ((size_t)bh * S_ + m0) * D_;
        for (int idx = tid; idx < M_ * D_; idx += NT)
            obase[idx] = ((float*)Osh)[idx];
    }
}

extern "C" void kernel_launch(void* const* d_in, const int* in_sizes, int n_in,
                              void* d_out, int out_size, void* d_ws, size_t ws_size,
                              hipStream_t stream) {
    const float* Q    = (const float*)d_in[0];
    const float* K    = (const float*)d_in[1];
    const float* V    = (const float*)d_in[2];
    const int*   mask = (const int*)d_in[3];

    float* out  = (float*)d_out;                          // [B,H,S,D] first
    float* attn = out + (size_t)B_ * H_ * S_ * D_;        // then [B,H,S,S]

    const int nblocks = B_ * H_ * (S_ / M_);              // 4096 row-blocks
    hipLaunchKernelGGL(attn_fused_kernel, dim3(nblocks), dim3(NT), 0, stream,
                       Q, K, V, mask, out, attn);
}
